// mLSTMblock_21534966022918
// MI455X (gfx1250) — matrix operand order repaired and checked by reading, not compiled, not run on hardware
//
#include <hip/hip_runtime.h>

// ---------------------------------------------------------------------------
// mLSTM block (two branches + bidirectional cross attention) for gfx1250.
// Heavy ops (linears, block-diagonal matmuls, causal conv, attention) run on
// V_WMMA_F32_16X16X32_F16 with fp32 accumulation, double-buffered LDS tiles.
// Conv weights/activations are pre-converted to f16 to halve hot-loop bytes.
// ---------------------------------------------------------------------------

typedef __attribute__((ext_vector_type(16))) _Float16 v16h;
typedef __attribute__((ext_vector_type(8)))  float    v8f;

// Fixed problem dims from the reference
constexpr int DIM_B  = 2;
constexpr int DIM_S  = 2048;
constexpr int DIM_D  = 512;
constexpr int DIM_H  = 1024;
constexpr int KCONV  = 51;
constexpr int NHEAD  = 2;
constexpr int DHEAD  = 64;
constexpr int NBS    = DIM_B * DIM_S;   // 4096 rows

// ---------------------------------------------------------------------------
// Device helpers
// ---------------------------------------------------------------------------
__device__ __forceinline__ float act_apply(float v, int act) {
  if (act == 1) return v / (1.f + __expf(-v));    // silu
  if (act == 2) return 1.f / (1.f + __expf(-v));  // sigmoid
  return v;
}

// One 64(M)x64(N)x32(K) step: each of the 4 waves owns a 16x64 strip of C.
// Fragment packing follows the CDNA5 16-bit A (16x32) / B (32x16) layouts.
__device__ __forceinline__ void mma_64x64_step(const _Float16 (*As)[34],
                                               const _Float16 (*Bs)[34],
                                               int wave, int lane, v8f* acc)
{
  const int hl = lane >> 4;        // lane group (0: lanes 0-15, 1: 16-31)
  const int lr = lane & 15;
  v16h af;
#pragma unroll
  for (int e = 0; e < 16; ++e) {
    // A 16x32 f16 layout: K = (e/8)*16 + ((e/2)%4)*2 + (e%2) + group*8
    const int kl = ((e >> 3) << 4) + (((e >> 1) & 3) << 1) + (e & 1) + (hl << 3);
    af[e] = As[wave * 16 + lr][kl];
  }
#pragma unroll
  for (int nt = 0; nt < 4; ++nt) {
    v16h bf;
#pragma unroll
    for (int e = 0; e < 16; ++e)          // B 32x16: lane=N, K = (e/8)*16 + group*8 + e%8
      bf[e] = Bs[nt * 16 + lr][((e >> 3) << 4) + (hl << 3) + (e & 7)];
    acc[nt] = __builtin_amdgcn_wmma_f32_16x16x32_f16(
        false, af, false, bf, (short)0, acc[nt], false, false);
  }
}

// ---------------------------------------------------------------------------
// Generic WMMA GEMM, double-buffered LDS staging:
//   C[m,n] = act( alpha * sum_k A[m*am + k*ak] * W[n*wn + k*wk] + bias[n] )
// Optional f16 mirror of C (for downstream f16 consumers).
// ---------------------------------------------------------------------------
__global__ __launch_bounds__(128)
void wmma_gemm_kernel(const float* __restrict__ A, long am, long ak,
                      const float* __restrict__ W, long wn, long wk,
                      const float* __restrict__ bias,
                      float* __restrict__ C, long ldc,
                      _Float16* __restrict__ C16,
                      int M, int N, int K, float alpha, int act)
{
  __shared__ _Float16 As[2][64][34];
  __shared__ _Float16 Bs[2][64][34];
  const int tid  = threadIdx.x;
  const int wave = tid >> 5;
  const int lane = tid & 31;
  const int m0 = blockIdx.y * 64;
  const int n0 = blockIdx.x * 64;

  v8f acc[4];
#pragma unroll
  for (int t = 0; t < 4; ++t)
#pragma unroll
    for (int e = 0; e < 8; ++e) acc[t][e] = 0.f;

  float ra[16], rw[16];
  auto load_tile = [&](int k0) {
    // speculative prefetch one tile ahead of this one
    __builtin_prefetch(&A[(long)(m0 + (tid & 63)) * am + (long)(k0 + 32) * ak], 0, 1);
    __builtin_prefetch(&W[(long)(n0 + (tid & 63)) * wn + (long)(k0 + 32) * wk], 0, 1);
#pragma unroll
    for (int j = 0; j < 16; ++j) {
      const int i = tid + j * 128;
      const int m = i >> 5, kk = i & 31;
      const int gk = k0 + kk;
      const int gm = m0 + m;
      const int gn = n0 + m;
      ra[j] = (gm < M && gk < K) ? A[(long)gm * am + (long)gk * ak] : 0.f;
      rw[j] = (gn < N && gk < K) ? W[(long)gn * wn + (long)gk * wk] : 0.f;
    }
  };
  auto store_tile = [&](int buf) {
#pragma unroll
    for (int j = 0; j < 16; ++j) {
      const int i = tid + j * 128;
      const int m = i >> 5, kk = i & 31;
      As[buf][m][kk] = (_Float16)ra[j];
      Bs[buf][m][kk] = (_Float16)rw[j];
    }
  };

  const int nk = (K + 31) >> 5;
  load_tile(0);
  store_tile(0);
  for (int t = 0; t < nk; ++t) {
    __syncthreads();                       // buffer t&1 fully staged
    if (t + 1 < nk) load_tile((t + 1) << 5);   // overlap next tile's loads
    mma_64x64_step(As[t & 1], Bs[t & 1], wave, lane, acc);
    if (t + 1 < nk) store_tile((t + 1) & 1);   // other buffer: no hazard
  }

  const int hl = lane >> 4, lr = lane & 15;
#pragma unroll
  for (int nt = 0; nt < 4; ++nt) {
    const int gn = n0 + nt * 16 + lr;
    if (gn >= N) continue;
    const float bv = bias ? bias[gn] : 0.f;
#pragma unroll
    for (int r = 0; r < 8; ++r) {
      const int gm = m0 + wave * 16 + hl * 8 + r;   // C/D f32 VGPR layout
      if (gm >= M) continue;
      const float v = act_apply(acc[nt][r] * alpha + bv, act);
      C[(long)gm * ldc + gn] = v;
      if (C16) C16[(long)gm * ldc + gn] = (_Float16)v;
    }
  }
}

// ---------------------------------------------------------------------------
// Conv weight repack: fp32 [co][ci][k] -> f16 [k][co][ci] (ci contiguous).
// ---------------------------------------------------------------------------
__global__ __launch_bounds__(256)
void convw_to_f16_kernel(const float* __restrict__ w, _Float16* __restrict__ o)
{
  const long t = (long)blockIdx.x * 256 + threadIdx.x;
  const long total = (long)DIM_H * DIM_H * KCONV;
  if (t >= total) return;
  const int ci = (int)(t & (DIM_H - 1));
  const long r = t >> 10;
  const int co = (int)(r & (DIM_H - 1));
  const int k  = (int)(r >> 10);
  o[t] = (_Float16)w[((long)co * DIM_H + ci) * KCONV + k];   // write coalesced
}

// ---------------------------------------------------------------------------
// Causal conv (H->H, K=51 taps) as K accumulated WMMA GEMMs, fused bias+SiLU.
// X16: f16 [B,S,H], W16: f16 [k][co][ci], Y: fp32 [B,S,H].
// Double-buffered LDS, packed b32 staging (2 halves per load/store).
// ---------------------------------------------------------------------------
__global__ __launch_bounds__(128)
void wmma_conv_silu_kernel(const _Float16* __restrict__ X16,
                           const _Float16* __restrict__ W16,
                           const float* __restrict__ bias,
                           float* __restrict__ Y)
{
  __shared__ _Float16 As[2][64][34];
  __shared__ _Float16 Bs[2][64][34];
  const int tid  = threadIdx.x;
  const int wave = tid >> 5;
  const int lane = tid & 31;
  const int t0 = blockIdx.x * 64;   // time tile
  const int b  = blockIdx.y;        // batch
  const int n0 = blockIdx.z * 64;   // out-channel tile

  v8f acc[4];
#pragma unroll
  for (int t = 0; t < 4; ++t)
#pragma unroll
    for (int e = 0; e < 8; ++e) acc[t][e] = 0.f;

  unsigned int ra[8], rb[8];
  auto load_tile = [&](int it) {
    const int kk = it >> 5;                 // conv tap
    const int c0 = (it & 31) << 5;          // ci tile base
#pragma unroll
    for (int j = 0; j < 8; ++j) {
      const int p  = tid + j * 128;         // pair index (1024 pairs / tile)
      const int m  = p >> 4;                // tile row (time or co)
      const int cp = (p & 15) << 1;         // ci offset in halves (even)
      const int ts = t0 + m + kk - (KCONV - 1);   // causal left pad
      unsigned int va = 0u;
      if (ts >= 0 && ts < DIM_S)
        va = *(const unsigned int*)(X16 + ((long)b * DIM_S + ts) * DIM_H + c0 + cp);
      ra[j] = va;
      rb[j] = *(const unsigned int*)(W16 + ((long)kk * DIM_H + n0 + m) * DIM_H + c0 + cp);
    }
  };
  auto store_tile = [&](int buf) {
#pragma unroll
    for (int j = 0; j < 8; ++j) {
      const int p  = tid + j * 128;
      const int m  = p >> 4;
      const int cp = (p & 15) << 1;
      *(unsigned int*)(&As[buf][m][cp]) = ra[j];
      *(unsigned int*)(&Bs[buf][m][cp]) = rb[j];
    }
  };

  constexpr int NT = KCONV * (DIM_H / 32);   // 1632 K-steps
  load_tile(0);
  store_tile(0);
  for (int t = 0; t < NT; ++t) {
    __syncthreads();
    if (t + 1 < NT) load_tile(t + 1);
    mma_64x64_step(As[t & 1], Bs[t & 1], wave, lane, acc);
    if (t + 1 < NT) store_tile((t + 1) & 1);
  }

  const int hl = lane >> 4, lr = lane & 15;
#pragma unroll
  for (int nt = 0; nt < 4; ++nt) {
    const int gn = n0 + nt * 16 + lr;
    const float bv = bias[gn];
#pragma unroll
    for (int r = 0; r < 8; ++r) {
      const int gt = t0 + wave * 16 + hl * 8 + r;
      const float v = acc[nt][r] + bv;
      Y[((long)b * DIM_S + gt) * DIM_H + gn] = v / (1.f + __expf(-v)); // silu
    }
  }
}

// ---------------------------------------------------------------------------
// LayerNorm over last dim (one row per block). act: 0 none, 1 exp, 2 sigmoid.
// In-place safe (Y may equal X).
// ---------------------------------------------------------------------------
__global__ __launch_bounds__(256)
void ln_kernel(const float* __restrict__ X, float* __restrict__ Y,
               const float* __restrict__ g, const float* __restrict__ b,
               int Cc, int act)
{
  const long row = blockIdx.x;
  const float* x = X + row * Cc;
  float* y = Y + row * Cc;
  __shared__ float red[256];
  const int tid = threadIdx.x;

  float s = 0.f;
  for (int c = tid; c < Cc; c += 256) s += x[c];
  red[tid] = s; __syncthreads();
  for (int o = 128; o > 0; o >>= 1) { if (tid < o) red[tid] += red[tid + o]; __syncthreads(); }
  const float mean = red[0] / (float)Cc;
  __syncthreads();

  float v = 0.f;
  for (int c = tid; c < Cc; c += 256) { const float d = x[c] - mean; v += d * d; }
  red[tid] = v; __syncthreads();
  for (int o = 128; o > 0; o >>= 1) { if (tid < o) red[tid] += red[tid + o]; __syncthreads(); }
  const float inv = rsqrtf(red[0] / (float)Cc + 1e-5f);
  __syncthreads();

  for (int c = tid; c < Cc; c += 256) {
    float val = (x[c] - mean) * inv * g[c] + b[c];
    if (act == 1) val = __expf(val);
    else if (act == 2) val = 1.f / (1.f + __expf(-val));
    y[c] = val;
  }
}

// ---------------------------------------------------------------------------
// Softmax over rows / columns of [gridDim.y] square matrices of size n x n.
// ---------------------------------------------------------------------------
__global__ __launch_bounds__(256)
void softmax_row_kernel(const float* __restrict__ X, float* __restrict__ Y, int n)
{
  const long base = (long)blockIdx.y * n * n + (long)blockIdx.x * n;
  const float* x = X + base;
  float* y = Y + base;
  __shared__ float red[256];
  const int tid = threadIdx.x;
  float m = -3.402823e38f;
  for (int c = tid; c < n; c += 256) m = fmaxf(m, x[c]);
  red[tid] = m; __syncthreads();
  for (int o = 128; o > 0; o >>= 1) { if (tid < o) red[tid] = fmaxf(red[tid], red[tid + o]); __syncthreads(); }
  m = red[0]; __syncthreads();
  float s = 0.f;
  for (int c = tid; c < n; c += 256) s += __expf(x[c] - m);
  red[tid] = s; __syncthreads();
  for (int o = 128; o > 0; o >>= 1) { if (tid < o) red[tid] += red[tid + o]; __syncthreads(); }
  const float inv = 1.f / red[0]; __syncthreads();
  for (int c = tid; c < n; c += 256) y[c] = __expf(x[c] - m) * inv;
}

__global__ __launch_bounds__(256)
void softmax_col_kernel(const float* __restrict__ X, float* __restrict__ Y, int n)
{
  const long base = (long)blockIdx.y * n * n + blockIdx.x;  // column
  const float* x = X + base;
  float* y = Y + base;
  __shared__ float red[256];
  const int tid = threadIdx.x;
  float m = -3.402823e38f;
  for (int r = tid; r < n; r += 256) m = fmaxf(m, x[(long)r * n]);
  red[tid] = m; __syncthreads();
  for (int o = 128; o > 0; o >>= 1) { if (tid < o) red[tid] = fmaxf(red[tid], red[tid + o]); __syncthreads(); }
  m = red[0]; __syncthreads();
  float s = 0.f;
  for (int r = tid; r < n; r += 256) s += __expf(x[(long)r * n] - m);
  red[tid] = s; __syncthreads();
  for (int o = 128; o > 0; o >>= 1) { if (tid < o) red[tid] += red[tid + o]; __syncthreads(); }
  const float inv = 1.f / red[0]; __syncthreads();
  for (int r = tid; r < n; r += 256) y[(long)r * n] = __expf(x[(long)r * n] - m) * inv;
}

// ---------------------------------------------------------------------------
// Column mean: out[c] = mean over R rows of X[r*C + c]
// ---------------------------------------------------------------------------
__global__ void colmean_kernel(const float* __restrict__ X, float* __restrict__ out,
                               int R, int Cc)
{
  const int c = blockIdx.x * blockDim.x + threadIdx.x;
  if (c >= Cc) return;
  float s = 0.f;
  for (int r = 0; r < R; ++r) s += X[(long)r * Cc + c];
  out[c] = s / (float)R;
}

// ---------------------------------------------------------------------------
// Two-stage deterministic global max of q[i] * ntv[i & (H-1)]
// ---------------------------------------------------------------------------
__global__ __launch_bounds__(256)
void max_partial_kernel(const float* __restrict__ q, const float* __restrict__ ntv,
                        float* __restrict__ part, long n)
{
  __shared__ float red[256];
  float m = -3.402823e38f;
  for (long i = (long)blockIdx.x * 256 + threadIdx.x; i < n; i += 256L * gridDim.x)
    m = fmaxf(m, q[i] * ntv[i & (DIM_H - 1)]);
  red[threadIdx.x] = m; __syncthreads();
  for (int o = 128; o > 0; o >>= 1) {
    if ((int)threadIdx.x < o) red[threadIdx.x] = fmaxf(red[threadIdx.x], red[threadIdx.x + o]);
    __syncthreads();
  }
  if (threadIdx.x == 0) part[blockIdx.x] = red[0];
}

__global__ __launch_bounds__(256)
void max_final_kernel(const float* __restrict__ part, float* __restrict__ out, int n)
{
  __shared__ float red[256];
  float m = -3.402823e38f;
  for (int i = threadIdx.x; i < n; i += 256) m = fmaxf(m, part[i]);
  red[threadIdx.x] = m; __syncthreads();
  for (int o = 128; o > 0; o >>= 1) {
    if ((int)threadIdx.x < o) red[threadIdx.x] = fmaxf(red[threadIdx.x], red[threadIdx.x + o]);
    __syncthreads();
  }
  if (threadIdx.x == 0) out[0] = red[0];
}

// ---------------------------------------------------------------------------
// Elementwise glue
// ---------------------------------------------------------------------------
__global__ void ew_mix_kernel(const float* a, const float* b, float* y, long n) {
  const long i = (long)blockIdx.x * blockDim.x + threadIdx.x;
  if (i < n) y[i] = 0.9f * a[i] + 0.1f * b[i];
}
__global__ void ew_ctnt_kernel(const float* iv, const float* vv, const float* kv,
                               float* ct, float* nt, long n) {
  const long i = (long)blockIdx.x * blockDim.x + threadIdx.x;
  if (i < n) { const float ii = iv[i], kk = kv[i]; ct[i] = ii * vv[i] * kk; nt[i] = ii * kk; }
}
__global__ void ew_ht_kernel(const float* o, const float* q, const float* ctv,
                             const float* mx, float* y, long n) {
  const long i = (long)blockIdx.x * blockDim.x + threadIdx.x;
  if (i < n) y[i] = o[i] * (ctv[i & (DIM_H - 1)] * q[i] / mx[0]);
}
__global__ void ew_add_kernel(const float* a, const float* b, float* y, long n) {
  const long i = (long)blockIdx.x * blockDim.x + threadIdx.x;
  if (i < n) y[i] = a[i] + b[i];
}
__global__ void ew_mul_kernel(const float* a, const float* b, float* y, long n) {
  const long i = (long)blockIdx.x * blockDim.x + threadIdx.x;
  if (i < n) y[i] = a[i] * b[i];
}

// ---------------------------------------------------------------------------
// Host-side launch helpers
// ---------------------------------------------------------------------------
static void gemm(hipStream_t st, const float* A, long am, long ak,
                 const float* W, long wn, long wk, const float* bias,
                 float* C, long ldc, int M, int N, int K,
                 float alpha = 1.f, int act = 0, _Float16* C16 = nullptr)
{
  dim3 g((unsigned)((N + 63) / 64), (unsigned)((M + 63) / 64));
  wmma_gemm_kernel<<<g, dim3(128), 0, st>>>(A, am, ak, W, wn, wk, bias,
                                            C, ldc, C16, M, N, K, alpha, act);
}
static void lnrun(hipStream_t st, const float* X, float* Y,
                  const float* g, const float* b, int rows, int Cc, int act)
{
  ln_kernel<<<dim3((unsigned)rows), dim3(256), 0, st>>>(X, Y, g, b, Cc, act);
}
static inline dim3 ew_grid(long n) { return dim3((unsigned)((n + 255) / 256)); }

// ---------------------------------------------------------------------------
// kernel_launch
// ---------------------------------------------------------------------------
extern "C" void kernel_launch(void* const* d_in, const int* in_sizes, int n_in,
                              void* d_out, int out_size, void* d_ws, size_t ws_size,
                              hipStream_t stream)
{
  (void)in_sizes; (void)n_in; (void)out_size; (void)ws_size;

  // Input flattening assumption: setup_inputs() dict insertion order, recursive.
  //   [0]=x_1, [1]=x_2, [2..37]=b1 (36 tensors), [38..73]=b2, [74..81]=bca.
  const float* x_in[2] = { (const float*)d_in[0], (const float*)d_in[1] };
  auto bp = [&](int branch, int idx) -> const float* {
    return (const float*)d_in[2 + branch * 36 + idx];
  };
  auto cp = [&](int idx) -> const float* {
    return (const float*)d_in[2 + 72 + idx];
  };
  // Branch param indices:
  //  0 ln_g 1 ln_b 2 left_w 3 left_b 4 right_w 5 right_b 6 conv_w 7 conv_b
  //  8 lskip_w 9 lskip_b 10 wq_w 11 wq_b 12 wk_w 13 wk_b 14 wv_w 15 wv_b
  // 16 ig_w 17 ig_b 18 og_w 19 og_b 20 lni_g 21 lni_b 22 lno_g 23 lno_b
  // 24 lnc_g 25 lnc_b 26 lnn_g 27 lnn_b 28 gn_g 29 gn_b 30 lnout_g 31 lnout_b
  // 32 proj_w 33 proj_b 34 lnproj_g 35 lnproj_b
  // bca: 0 to_qk 1 c_to_qk 2 to_v 3 c_to_v 4 to_out_w 5 to_out_b 6 c_to_out_w 7 c_to_out_b

  const size_t BSH   = (size_t)NBS * DIM_H;
  const size_t BSD   = (size_t)NBS * DIM_D;
  const size_t BS128 = (size_t)NBS * NHEAD * DHEAD;
  const size_t SIMSZ = (size_t)DIM_B * NHEAD * DIM_S * DIM_S;
  const size_t WCONV = (size_t)KCONV * DIM_H * DIM_H;   // f16 elements

  float* ws = (float*)d_ws;
  size_t off = 0;
  auto alloc = [&](size_t n) -> float* {        // n floats
    float* p = ws + off;
    off += (n + 255) & ~(size_t)255;
    return p;
  };
  auto alloc_h = [&](size_t n) -> _Float16* {   // n halves
    return (_Float16*)alloc((n + 1) / 2);
  };

  float* xn   = alloc(BSD);   // shared across branches (sequential pre)
  float* left = alloc(BSH);   // shared across branches
  _Float16* left16 = alloc_h(BSH);   // f16 mirror of `left` (conv input)
  _Float16* wt16   = alloc_h(WCONV); // f16 repacked conv weights [k][co][ci]
  float *right[2], *ll[2], *lskip[2], *qg[2], *kg[2], *vg[2], *ipre[2], *ir[2];
  for (int i = 0; i < 2; ++i) {
    right[i] = alloc(BSH); ll[i] = alloc(BSH); lskip[i] = alloc(BSH);
    qg[i] = alloc(BSH); kg[i] = alloc(BSH); vg[i] = alloc(BSH);
    ipre[i] = alloc(BSH); ir[i] = alloc(BSH);
  }
  float* qk    = alloc(BS128);
  float* vv    = alloc(BS128);
  float* cqk   = alloc(BS128);
  float* cv    = alloc(BS128);
  float* sim   = alloc(SIMSZ);   // becomes cattn in-place
  float* attn  = alloc(SIMSZ);
  float* outm  = alloc(BS128);
  float* coutm = alloc(BS128);
  float* iexp  = alloc(BSH);     // shared post temporaries (sequential post)
  float* obuf  = alloc(BSH);
  float* ctf   = alloc(BSH);
  float* ntf   = alloc(BSH);
  float* tmp   = alloc(BSH);
  float* projd = alloc(BSD);
  float* ctvec = alloc(DIM_H);
  float* ntvec = alloc(DIM_H);
  float* part  = alloc(512);
  float* maxb  = alloc(16);

  float* outp = (float*)d_out;   // [out1 | out2], each [B,S,D] fp32

  // ------------------------------ pre (both branches) ----------------------
  for (int i = 0; i < 2; ++i) {
    lnrun(stream, x_in[i], xn, bp(i,0), bp(i,1), NBS, DIM_D, 0);
    // left / right projections D -> H (right fused silu); left also emits f16
    gemm(stream, xn, DIM_D, 1, bp(i,2), DIM_D, 1, bp(i,3), left,     DIM_H, NBS, DIM_H, DIM_D, 1.f, 0, left16);
    gemm(stream, xn, DIM_D, 1, bp(i,4), DIM_D, 1, bp(i,5), right[i], DIM_H, NBS, DIM_H, DIM_D, 1.f, 1);
    // repack conv weights fp32 [co][ci][k] -> f16 [k][co][ci]
    convw_to_f16_kernel<<<dim3((unsigned)((WCONV + 255) / 256)), dim3(256), 0, stream>>>(
        bp(i,6), wt16);
    // causal conv + bias + silu -> ll (f16 inputs, fp32 WMMA accumulate)
    wmma_conv_silu_kernel<<<dim3(DIM_S/64, DIM_B, DIM_H/64), dim3(128), 0, stream>>>(
        left16, wt16, bp(i,7), ll[i]);
    // l_skip = lin(ll)
    gemm(stream, ll[i], DIM_H, 1, bp(i,8), DIM_H, 1, bp(i,9), lskip[i], DIM_H, NBS, DIM_H, DIM_H, 1.f, 0);
    // block-diagonal q (ll), k (ll), v (left): 8 blocks of 128x128
    for (int d = 0; d < 8; ++d) {
      const long wo = (long)d * 128 * 128;
      gemm(stream, ll[i] + d*128, DIM_H, 1, bp(i,10)+wo, 128, 1, bp(i,11)+d*128, qg[i]+d*128, DIM_H, NBS, 128, 128, 1.f, 0);
      gemm(stream, ll[i] + d*128, DIM_H, 1, bp(i,12)+wo, 128, 1, bp(i,13)+d*128, kg[i]+d*128, DIM_H, NBS, 128, 128, 1.f, 0);
      gemm(stream, left  + d*128, DIM_H, 1, bp(i,14)+wo, 128, 1, bp(i,15)+d*128, vg[i]+d*128, DIM_H, NBS, 128, 128, 1.f, 0);
    }
    // i_pre = lin(ll, ig)
    gemm(stream, ll[i], DIM_H, 1, bp(i,16), DIM_H, 1, bp(i,17), ipre[i], DIM_H, NBS, DIM_H, DIM_H, 1.f, 0);
  }

  // ------------------------------ BCA --------------------------------------
  const int IN = NHEAD * DHEAD;   // 128
  gemm(stream, ipre[0], DIM_H, 1, cp(0), DIM_H, 1, nullptr, qk,  IN, NBS, IN, DIM_H, 1.f, 0);
  gemm(stream, ipre[0], DIM_H, 1, cp(2), DIM_H, 1, nullptr, vv,  IN, NBS, IN, DIM_H, 1.f, 0);
  gemm(stream, ipre[1], DIM_H, 1, cp(1), DIM_H, 1, nullptr, cqk, IN, NBS, IN, DIM_H, 1.f, 0);
  gemm(stream, ipre[1], DIM_H, 1, cp(3), DIM_H, 1, nullptr, cv,  IN, NBS, IN, DIM_H, 1.f, 0);

  const float scale = 0.125f;  // DHEAD^-0.5
  for (int b = 0; b < DIM_B; ++b)
    for (int h = 0; h < NHEAD; ++h) {
      const long moff = ((long)b * NHEAD + h) * (long)DIM_S * DIM_S;
      const long hoff = (long)b * DIM_S * IN + h * DHEAD;
      // sim[i,j] = scale * qk[i,:] . cqk[j,:]
      gemm(stream, qk + hoff, IN, 1, cqk + hoff, IN, 1, nullptr,
           sim + moff, DIM_S, DIM_S, DIM_S, DHEAD, scale, 0);
    }
  softmax_row_kernel<<<dim3(DIM_S, DIM_B * NHEAD), dim3(256), 0, stream>>>(sim, attn, DIM_S);
  softmax_col_kernel<<<dim3(DIM_S, DIM_B * NHEAD), dim3(256), 0, stream>>>(sim, sim,  DIM_S); // sim := cattn
  for (int b = 0; b < DIM_B; ++b)
    for (int h = 0; h < NHEAD; ++h) {
      const long moff = ((long)b * NHEAD + h) * (long)DIM_S * DIM_S;
      const long hoff = (long)b * DIM_S * IN + h * DHEAD;
      // out[i,d]  = sum_j attn[i,j]  * cv[j,d]
      gemm(stream, attn + moff, DIM_S, 1, cv + hoff, 1, IN, nullptr,
           outm + hoff, IN, DIM_S, DHEAD, DIM_S, 1.f, 0);
      // cout[j,d] = sum_i cattn[i,j] * v[i,d]
      gemm(stream, sim + moff, 1, DIM_S, vv + hoff, 1, IN, nullptr,
           coutm + hoff, IN, DIM_S, DHEAD, DIM_S, 1.f, 0);
    }
  gemm(stream, outm,  IN, 1, cp(4), IN, 1, cp(5), ir[0], DIM_H, NBS, DIM_H, IN, 1.f, 0);
  gemm(stream, coutm, IN, 1, cp(6), IN, 1, cp(7), ir[1], DIM_H, NBS, DIM_H, IN, 1.f, 0);
  // i_mixed = 0.9*i_pre + 0.1*i_r  (in place in ir[])
  ew_mix_kernel<<<ew_grid(BSH), dim3(256), 0, stream>>>(ipre[0], ir[0], ir[0], (long)BSH);
  ew_mix_kernel<<<ew_grid(BSH), dim3(256), 0, stream>>>(ipre[1], ir[1], ir[1], (long)BSH);

  // ------------------------------ post (both branches) ---------------------
  for (int i = 0; i < 2; ++i) {
    lnrun(stream, ir[i], iexp, bp(i,20), bp(i,21), NBS, DIM_H, 1);         // i = exp(LN(i_mixed))
    gemm(stream, ll[i], DIM_H, 1, bp(i,18), DIM_H, 1, bp(i,19), obuf, DIM_H, NBS, DIM_H, DIM_H, 1.f, 0);
    lnrun(stream, obuf, obuf, bp(i,22), bp(i,23), NBS, DIM_H, 2);          // o = sigmoid(LN(...))
    ew_ctnt_kernel<<<ew_grid(BSH), dim3(256), 0, stream>>>(iexp, vg[i], kg[i], ctf, ntf, (long)BSH);
    lnrun(stream, ctf, ctf, bp(i,24), bp(i,25), NBS, DIM_H, 0);
    lnrun(stream, ntf, ntf, bp(i,26), bp(i,27), NBS, DIM_H, 0);
    colmean_kernel<<<dim3((DIM_H + 255) / 256), dim3(256), 0, stream>>>(ctf, ctvec, NBS, DIM_H);
    colmean_kernel<<<dim3((DIM_H + 255) / 256), dim3(256), 0, stream>>>(ntf, ntvec, NBS, DIM_H);
    max_partial_kernel<<<dim3(256), dim3(256), 0, stream>>>(qg[i], ntvec, part, (long)BSH);
    max_final_kernel<<<dim3(1), dim3(256), 0, stream>>>(part, maxb, 256);
    ew_ht_kernel<<<ew_grid(BSH), dim3(256), 0, stream>>>(obuf, qg[i], ctvec, maxb, tmp, (long)BSH);
    ew_add_kernel<<<ew_grid(BSH), dim3(256), 0, stream>>>(tmp, lskip[i], tmp, (long)BSH);
    lnrun(stream, tmp, tmp, bp(i,28), bp(i,29), NBS, DIM_H, 0);            // left_o
    ew_mul_kernel<<<ew_grid(BSH), dim3(256), 0, stream>>>(tmp, right[i], tmp, (long)BSH);
    lnrun(stream, tmp, tmp, bp(i,30), bp(i,31), NBS, DIM_H, 0);            // out
    gemm(stream, tmp, DIM_H, 1, bp(i,32), DIM_H, 1, bp(i,33), projd, DIM_D, NBS, DIM_D, DIM_H, 1.f, 0);
    lnrun(stream, projd, outp + (size_t)i * BSD, bp(i,34), bp(i,35), NBS, DIM_D, 0);
  }
}
